// GPTModel_55972013802258
// MI455X (gfx1250) — matrix operand order repaired and checked by reading, not compiled, not run on hardware
//
#include <hip/hip_runtime.h>
#include <stdint.h>

// ---------------- CDNA5 types ----------------
typedef __attribute__((ext_vector_type(16))) __bf16          v16bf;
typedef __attribute__((ext_vector_type(8)))  float           v8f;
typedef __attribute__((ext_vector_type(16))) unsigned short  v16u;
typedef __attribute__((ext_vector_type(8)))  unsigned short  v8u;
typedef __attribute__((ext_vector_type(4)))  unsigned int    v4u;
typedef __attribute__((ext_vector_type(8)))  int             v8i;
typedef __attribute__((ext_vector_type(4)))  int             v4i;

#define S_LEN   2048
#define EMB     768
#define NHEAD   12
#define HDIM    64
#define VOCAB   50257
#define FFN_DIM 3072

#if defined(__has_builtin)
#if __has_builtin(__builtin_amdgcn_tensor_load_to_lds)
#define HAVE_TDM 1
#endif
#endif

// f32 -> bf16 (round to nearest even)
static __device__ __forceinline__ unsigned short f2bf(float f) {
  union { float f; unsigned u; } cv; cv.f = f;
  unsigned u = cv.u;
  unsigned r = (u + 0x7FFFu + ((u >> 16) & 1u)) >> 16;
  return (unsigned short)r;
}

static __device__ __forceinline__ v8f wmma_bf16(v16bf a, v16bf b, v8f c) {
  return __builtin_amdgcn_wmma_f32_16x16x32_bf16(false, a, false, b, (short)0, c, false, false);
}

// Build a 16-element bf16 fragment from two contiguous 16B chunks.
static __device__ __forceinline__ v16bf load_frag16(const unsigned short* p0,
                                                    const unsigned short* p1) {
  v8u lo = *(const v8u*)p0;
  v8u hi = *(const v8u*)p1;
  v16u cat = __builtin_shufflevector(lo, hi, 0,1,2,3,4,5,6,7,8,9,10,11,12,13,14,15);
  union { v16u u; v16bf b; } cv; cv.u = cat;
  return cv.b;
}

#ifdef HAVE_TDM
// Tensor Data Mover: 2D bf16 tile (tile_d1 rows x tile_d0 elems) from global to LDS.
// D# built per CDNA5 ISA 8.3/8.4: group0 = {count, lds_addr, global_addr, type=2},
// group1 = {data_size=2B, LDS pad config, tensor dims, tile dims, dim0 stride}.
static __device__ __forceinline__ void tdm_load_2d(
    unsigned lds_off, const void* gptr,
    unsigned tensor_d0, unsigned tensor_d1,
    unsigned tile_d0, unsigned tile_d1,
    unsigned long long stride_d0,           // elements
    unsigned pad_enable, unsigned pad_interval, unsigned pad_amount) {
  unsigned long long ga = (unsigned long long)(uintptr_t)gptr;
  v4u g0;
  g0[0] = 1u;                                            // count=1, user mode
  g0[1] = lds_off;                                       // lds_addr [63:32]
  g0[2] = (unsigned)ga;                                  // global_addr low
  g0[3] = (unsigned)((ga >> 32) & 0x01FFFFFFu) | 0x80000000u;  // addr[56:32] | type=2
  v8i g1;
  g1[0] = (int)((1u << 16)                               // data_size = 2 bytes
                | (pad_enable << 20)
                | (pad_interval << 22)
                | (pad_amount << 25));
  g1[1] = (int)((tensor_d0 & 0xFFFFu) << 16);            // tensor_dim0 [79:48] lo
  g1[2] = (int)(((tensor_d0 >> 16) & 0xFFFFu)            // tensor_dim0 hi
                | ((tensor_d1 & 0xFFFFu) << 16));        // tensor_dim1 lo
  g1[3] = (int)(((tensor_d1 >> 16) & 0xFFFFu)            // tensor_dim1 hi
                | ((tile_d0 & 0xFFFFu) << 16));          // tile_dim0 [127:112]
  g1[4] = (int)(tile_d1 & 0xFFFFu);                      // tile_dim1 (tile_dim2=0)
  g1[5] = (int)(unsigned)(stride_d0 & 0xFFFFFFFFull);    // dim0 stride lo
  g1[6] = (int)(unsigned)((stride_d0 >> 32) & 0xFFFFull);// dim0 stride hi
  g1[7] = 0;                                             // dim1 stride (2D: unused)
  v4i z4 = {0, 0, 0, 0};
#if __clang_major__ >= 23
  v8i z8 = {0, 0, 0, 0, 0, 0, 0, 0};
  __builtin_amdgcn_tensor_load_to_lds(g0, g1, z4, z4, z8, 0);
#else
  __builtin_amdgcn_tensor_load_to_lds(g0, g1, z4, z4, 0);
#endif
}

static __device__ __forceinline__ void tdm_wait0() {
#if __has_builtin(__builtin_amdgcn_s_wait_tensorcnt)
  __builtin_amdgcn_s_wait_tensorcnt(0);
#else
  asm volatile("s_wait_tensorcnt 0" ::: "memory");
#endif
}
#endif  // HAVE_TDM

// ---------------- elementwise kernels ----------------
__global__ void cvt_kernel(const float* __restrict__ in, unsigned short* __restrict__ out, int n) {
  int i = blockIdx.x * 256 + threadIdx.x;
  if (i < n) out[i] = f2bf(in[i]);
}

__global__ void embed_kernel(const int* __restrict__ X, const float* __restrict__ tok,
                             const float* __restrict__ pos, float* __restrict__ h) {
  int idx = blockIdx.x * 256 + threadIdx.x;
  if (idx >= S_LEN * EMB) return;
  int s = idx / EMB, e = idx - s * EMB;
  int tk = X[s];
  h[idx] = tok[(size_t)tk * EMB + e] + pos[idx];
}

// block-per-row LayerNorm over E=768; writes bf16 (+ optional f32)
__global__ __launch_bounds__(256) void ln_kernel(const float* __restrict__ x,
    const float* __restrict__ g, const float* __restrict__ b,
    unsigned short* __restrict__ outb, float* __restrict__ outf) {
  __shared__ float red[256];
  const int row = blockIdx.x, t = threadIdx.x;
  const float* xr = x + (size_t)row * EMB;
  float v0 = xr[t], v1 = xr[t + 256], v2 = xr[t + 512];
  red[t] = v0 + v1 + v2;
  __syncthreads();
  for (int st = 128; st > 0; st >>= 1) { if (t < st) red[t] += red[t + st]; __syncthreads(); }
  const float mu = red[0] * (1.0f / EMB);
  __syncthreads();
  float d0 = v0 - mu, d1 = v1 - mu, d2 = v2 - mu;
  red[t] = d0 * d0 + d1 * d1 + d2 * d2;
  __syncthreads();
  for (int st = 128; st > 0; st >>= 1) { if (t < st) red[t] += red[t + st]; __syncthreads(); }
  const float inv = rsqrtf(red[0] * (1.0f / EMB) + 1e-5f);
  float vv[3] = { v0, v1, v2 };
  #pragma unroll
  for (int j = 0; j < 3; ++j) {
    const int e = t + j * 256;
    const float val = (vv[j] - mu) * inv * g[e] + b[e];
    outb[(size_t)row * EMB + e] = f2bf(val);
    if (outf) outf[(size_t)row * EMB + e] = val;
  }
}

// ---------------- WMMA GEMM: C[M,N] = epi(A_bf16[M,K] @ W_bf16[K,N] + bias) ----------------
// EPI: 0 = store bf16, 1 = exact GELU -> bf16, 2 = residual add -> f32
// Tiles staged by the Tensor Data Mover (wave 0 issues, all waves consume).
template<int EPI>
__global__ __launch_bounds__(256) void gemm_kernel(const unsigned short* __restrict__ A,
    const unsigned short* __restrict__ W, const float* __restrict__ bias,
    const float* __restrict__ resid, unsigned short* __restrict__ outb,
    float* __restrict__ outf, int M, int N, int K) {
  __shared__ unsigned short Als[128 * 40];  // pitch 40 elems (80B): conflict-free, 16B aligned
  __shared__ unsigned short Bls[32 * 72];   // pitch 72 elems

  const int t = threadIdx.x;
  const int lane = t & 31;
  const int wave = t >> 5;
  const int wm = wave & 3, wn = wave >> 2;           // 4x2 wave grid -> 128x64 tile
  const int m0 = blockIdx.y * 128, n0 = blockIdx.x * 64;

  v8f c[2][2] = {};

  const int mloc = lane & 15;
  const int kb2  = (lane & 16) ? 8 : 0;              // A-frag K base
  const int koff = (lane & 16) ? 16 : 0;             // unused; B-frag also uses kb2
  const int h8   = (lane & 16) ? 8 : 0;              // C-frag row base

#ifdef HAVE_TDM
  const unsigned ldsA = (unsigned)(uintptr_t)Als;
  const unsigned ldsB = (unsigned)(uintptr_t)Bls;
#else
  const int arow_l = t >> 1, ach = (t & 1) << 4;     // A tile: 128 rows x 32 cols
  const int brow_l = t >> 3, bch = (t & 7) << 3;     // B tile: 32 rows x 64 cols
#endif

  for (int k0 = 0; k0 < K; k0 += 32) {
#ifdef HAVE_TDM
    if (wave == 0) {
      // A tile: 128 rows x 32 bf16; LDS rows padded 64B+16B -> pitch 40 elems.
      // pad_interval code 3 = 16 DWORDs (64B), pad_amount code 3 = 4 DWORDs (16B).
      tdm_load_2d(ldsA, A + (size_t)m0 * K + k0, (unsigned)K, (unsigned)(M - m0),
                  32u, 128u, (unsigned long long)K, 1u, 3u, 3u);
      // W tile: 32 rows x 64 bf16; LDS rows padded 128B+16B -> pitch 72 elems.
      // pad_interval code 4 = 32 DWORDs (128B), pad_amount code 3 = 4 DWORDs.
      tdm_load_2d(ldsB, W + (size_t)k0 * N + n0, (unsigned)(N - n0), (unsigned)(K - k0),
                  64u, 32u, (unsigned long long)N, 1u, 4u, 3u);
      tdm_wait0();
    }
#else
    const uint4* ga = (const uint4*)(A + (size_t)(m0 + arow_l) * K + k0 + ach);
    uint4* la = (uint4*)(Als + arow_l * 40 + ach);
    la[0] = ga[0]; la[1] = ga[1];
    uint4 wv = *(const uint4*)(W + (size_t)(k0 + brow_l) * N + n0 + bch);
    unsigned short* lb = Bls + brow_l * 72 + bch;
    *(uint2*)(lb)     = make_uint2(wv.x, wv.y);
    *(uint2*)(lb + 4) = make_uint2(wv.z, wv.w);
#endif
    __syncthreads();

    v16bf af[2], bfv[2];
    #pragma unroll
    for (int mi = 0; mi < 2; ++mi) {
      const unsigned short* base = Als + (wm * 32 + mi * 16 + mloc) * 40;
      af[mi] = load_frag16(base + kb2, base + 16 + kb2);
    }
    #pragma unroll
    for (int ni = 0; ni < 2; ++ni) {
      v16u bu;
      const int coln = wn * 32 + ni * 16 + mloc;
      #pragma unroll
      for (int e = 0; e < 16; ++e) bu[e] = Bls[(kb2 + (e & 7) + ((e >> 3) << 4)) * 72 + coln];
      union { v16u u; v16bf b; } cv; cv.u = bu;
      bfv[ni] = cv.b;
    }
    #pragma unroll
    for (int mi = 0; mi < 2; ++mi)
      #pragma unroll
      for (int ni = 0; ni < 2; ++ni)
        c[mi][ni] = wmma_bf16(af[mi], bfv[ni], c[mi][ni]);
    __syncthreads();
  }

  #pragma unroll
  for (int mi = 0; mi < 2; ++mi) {
    #pragma unroll
    for (int ni = 0; ni < 2; ++ni) {
      const int colg = n0 + wn * 32 + ni * 16 + mloc;
      const float bc = bias[colg];
      #pragma unroll
      for (int r = 0; r < 8; ++r) {
        const int rowg = m0 + wm * 32 + mi * 16 + r + h8;
        const size_t idx = (size_t)rowg * N + colg;
        float vv = c[mi][ni][r] + bc;
        if (EPI == 0) {
          outb[idx] = f2bf(vv);
        } else if (EPI == 1) {
          float gel = 0.5f * vv * (1.0f + erff(vv * 0.70710678118654752f));
          outb[idx] = f2bf(gel);
        } else {
          outf[idx] = resid[idx] + vv;
        }
      }
    }
  }
}

// ---------------- flash attention: one wave per (head, 16-query tile) ----------------
__global__ __launch_bounds__(128) void attn_kernel(const unsigned short* __restrict__ q,
    const unsigned short* __restrict__ k, const unsigned short* __restrict__ v,
    unsigned short* __restrict__ out) {
  __shared__ unsigned short Kt[4][32 * 64];
  __shared__ unsigned short Vt[4][32 * 64];
  __shared__ unsigned short Pt[4][16 * 32];

  const int lane = threadIdx.x & 31;
  const int wv_  = threadIdx.x >> 5;
  const int gw   = blockIdx.x * 4 + wv_;
  const int hh   = gw >> 7;               // 128 query tiles per head
  const int q0   = (gw & 127) << 4;

  unsigned short* Ktw = Kt[wv_];
  unsigned short* Vtw = Vt[wv_];
  unsigned short* Ptw = Pt[wv_];

  const int mloc = lane & 15;
  const int kb2  = (lane & 16) ? 8 : 0;
  const int koff = (lane & 16) ? 16 : 0;
  const int h8   = (lane & 16) ? 8 : 0;

  // Q fragments (16x64 split into two 16x32 A-fragments), loaded once
  const unsigned short* qrow = q + (size_t)(q0 + mloc) * EMB + hh * HDIM;
  v16bf qf0 = load_frag16(qrow + kb2,      qrow + 16 + kb2);
  v16bf qf1 = load_frag16(qrow + 32 + kb2, qrow + 48 + kb2);

  float mr[8], lr[8];
  #pragma unroll
  for (int r = 0; r < 8; ++r) { mr[r] = -1e30f; lr[r] = 0.0f; }
  v8f oacc[4] = {};

#ifdef HAVE_TDM
  const unsigned ldsK = (unsigned)(uintptr_t)Ktw;
  const unsigned ldsV = (unsigned)(uintptr_t)Vtw;
#endif

  for (int kb = 0; kb <= q0 + 15; kb += 32) {
#ifdef HAVE_TDM
    // Per-wave TDM DMA of the K,V tiles (32 keys x 64 dims, packed pitch 64).
    tdm_load_2d(ldsK, k + (size_t)kb * EMB + hh * HDIM,
                (unsigned)EMB, (unsigned)(S_LEN - kb), 64u, 32u,
                (unsigned long long)EMB, 0u, 0u, 0u);
    tdm_load_2d(ldsV, v + (size_t)kb * EMB + hh * HDIM,
                (unsigned)EMB, (unsigned)(S_LEN - kb), 64u, 32u,
                (unsigned long long)EMB, 0u, 0u, 0u);
    tdm_wait0();
#else
    const uint4* gk = (const uint4*)(k + (size_t)(kb + lane) * EMB + hh * HDIM);
    const uint4* gv = (const uint4*)(v + (size_t)(kb + lane) * EMB + hh * HDIM);
    uint4* lk = (uint4*)(Ktw + lane * 64);
    uint4* lv = (uint4*)(Vtw + lane * 64);
    #pragma unroll
    for (int i = 0; i < 8; ++i) { lk[i] = gk[i]; lv[i] = gv[i]; }
    asm volatile("s_wait_dscnt 0" ::: "memory");
#endif

    // scores S = Q K^T for key columns [0..15] and [16..31]
    v8f s0 = {}, s1 = {};
    {
      const unsigned short* kr0 = Ktw + (0  + mloc) * 64;
      const unsigned short* kr1 = Ktw + (16 + mloc) * 64;
      s0 = wmma_bf16(qf0, load_frag16(kr0 + kb2,      kr0 + 16 + kb2),      s0);
      s0 = wmma_bf16(qf1, load_frag16(kr0 + 32 + kb2, kr0 + 48 + kb2), s0);
      s1 = wmma_bf16(qf0, load_frag16(kr1 + kb2,      kr1 + 16 + kb2),      s1);
      s1 = wmma_bf16(qf1, load_frag16(kr1 + 32 + kb2, kr1 + 48 + kb2), s1);
    }

    // online softmax: per-row (16-lane half-wave) reductions
    #pragma unroll
    for (int r = 0; r < 8; ++r) {
      const int qrowi = q0 + r + h8;
      float a0 = s0[r] * 0.125f;                 // 1/sqrt(64)
      if (kb + mloc > qrowi) a0 = -1e30f;        // causal mask
      float a1 = s1[r] * 0.125f;
      if (kb + 16 + mloc > qrowi) a1 = -1e30f;
      float bm = fmaxf(a0, a1);
      bm = fmaxf(bm, __shfl_xor(bm, 8, 16));
      bm = fmaxf(bm, __shfl_xor(bm, 4, 16));
      bm = fmaxf(bm, __shfl_xor(bm, 2, 16));
      bm = fmaxf(bm, __shfl_xor(bm, 1, 16));
      const float mnew = fmaxf(mr[r], bm);
      const float al   = __expf(mr[r] - mnew);
      const float p0   = __expf(a0 - mnew);
      const float p1   = __expf(a1 - mnew);
      float rs = p0 + p1;
      rs += __shfl_xor(rs, 8, 16);
      rs += __shfl_xor(rs, 4, 16);
      rs += __shfl_xor(rs, 2, 16);
      rs += __shfl_xor(rs, 1, 16);
      lr[r] = lr[r] * al + rs;
      mr[r] = mnew;
      #pragma unroll
      for (int ni = 0; ni < 4; ++ni) oacc[ni][r] *= al;
      Ptw[(r + h8) * 32 + mloc]      = f2bf(p0);  // C-layout -> LDS
      Ptw[(r + h8) * 32 + 16 + mloc] = f2bf(p1);
    }
    asm volatile("s_wait_dscnt 0" ::: "memory");

    // reload P as 16x32 A-fragment; O += P @ V (4 WMMAs over N=64)
    const unsigned short* pr = Ptw + mloc * 32;
    v16bf pf = load_frag16(pr + kb2, pr + 16 + kb2);
    #pragma unroll
    for (int ni = 0; ni < 4; ++ni) {
      v16u bu;
      #pragma unroll
      for (int e = 0; e < 16; ++e) bu[e] = Vtw[(kb2 + (e & 7) + ((e >> 3) << 4)) * 64 + ni * 16 + mloc];
      union { v16u u; v16bf b; } cv; cv.u = bu;
      oacc[ni] = wmma_bf16(pf, cv.b, oacc[ni]);
    }
  }

  // normalize and store bf16 to [s, h*64+d]
  #pragma unroll
  for (int ni = 0; ni < 4; ++ni)
    #pragma unroll
    for (int r = 0; r < 8; ++r) {
      const size_t idx = (size_t)(q0 + r + h8) * EMB + hh * HDIM + ni * 16 + mloc;
      out[idx] = f2bf(oacc[ni][r] / lr[r]);
    }
}

// ---------------- last-token head GEMV (bandwidth bound) ----------------
__global__ __launch_bounds__(256) void head_kernel(const float* __restrict__ hlast,
    const float* __restrict__ W, const float* __restrict__ b, float* __restrict__ out) {
  __shared__ float hl[EMB];
  const int t = threadIdx.x;
  hl[t] = hlast[t]; hl[t + 256] = hlast[t + 256]; hl[t + 512] = hlast[t + 512];
  __syncthreads();
  const int vtx = blockIdx.x * 256 + t;
  if (vtx >= VOCAB) return;
  float acc = b[vtx];
  for (int e = 0; e < EMB; ++e) acc = fmaf(hl[e], W[(size_t)e * VOCAB + vtx], acc);
  out[vtx] = acc;
}

// ---------------- host orchestration ----------------
extern "C" void kernel_launch(void* const* d_in, const int* in_sizes, int n_in,
                              void* d_out, int out_size, void* d_ws, size_t ws_size,
                              hipStream_t stream) {
  (void)in_sizes; (void)n_in; (void)out_size; (void)ws_size;
  char* ws = (char*)d_ws;
  size_t off = 0;
  auto alloc = [&](size_t bytes) -> char* {
    off = (off + 255) & ~(size_t)255;
    char* p = ws + off; off += bytes; return p;
  };

  float* h  = (float*)alloc((size_t)S_LEN * EMB * 4);
  float* hf = (float*)alloc((size_t)S_LEN * EMB * 4);
  unsigned short* xb   = (unsigned short*)alloc((size_t)S_LEN * EMB * 2);
  unsigned short* qbuf = (unsigned short*)alloc((size_t)S_LEN * EMB * 2);
  unsigned short* kbuf = (unsigned short*)alloc((size_t)S_LEN * EMB * 2);
  unsigned short* vbuf = (unsigned short*)alloc((size_t)S_LEN * EMB * 2);
  unsigned short* abuf = (unsigned short*)alloc((size_t)S_LEN * EMB * 2);
  unsigned short* midb = (unsigned short*)alloc((size_t)S_LEN * FFN_DIM * 2);

  unsigned short *wq[4], *wk[4], *wvv[4], *wo[4], *wd1[4], *wd2[4];
  for (int b = 0; b < 4; ++b) {
    wq[b]  = (unsigned short*)alloc((size_t)EMB * EMB * 2);
    wk[b]  = (unsigned short*)alloc((size_t)EMB * EMB * 2);
    wvv[b] = (unsigned short*)alloc((size_t)EMB * EMB * 2);
    wo[b]  = (unsigned short*)alloc((size_t)EMB * EMB * 2);
    wd1[b] = (unsigned short*)alloc((size_t)EMB * FFN_DIM * 2);
    wd2[b] = (unsigned short*)alloc((size_t)EMB * FFN_DIM * 2);
  }

  // input order: 0:X 1:tok_emb 2:pos, then 16 tensors per block x4, then lnf_g,lnf_b,head_W,head_b
  const int nEE = EMB * EMB, nEF = EMB * FFN_DIM;
  for (int b = 0; b < 4; ++b) {
    const int base = 3 + b * 16;
    cvt_kernel<<<(nEE + 255) / 256, 256, 0, stream>>>((const float*)d_in[base + 2],  wq[b],  nEE);
    cvt_kernel<<<(nEE + 255) / 256, 256, 0, stream>>>((const float*)d_in[base + 4],  wk[b],  nEE);
    cvt_kernel<<<(nEE + 255) / 256, 256, 0, stream>>>((const float*)d_in[base + 6],  wvv[b], nEE);
    cvt_kernel<<<(nEE + 255) / 256, 256, 0, stream>>>((const float*)d_in[base + 8],  wo[b],  nEE);
    cvt_kernel<<<(nEF + 255) / 256, 256, 0, stream>>>((const float*)d_in[base + 12], wd1[b], nEF);
    cvt_kernel<<<(nEF + 255) / 256, 256, 0, stream>>>((const float*)d_in[base + 14], wd2[b], nEF);
  }

  embed_kernel<<<(S_LEN * EMB + 255) / 256, 256, 0, stream>>>(
      (const int*)d_in[0], (const float*)d_in[1], (const float*)d_in[2], h);

  const dim3 gE(EMB / 64, S_LEN / 128);       // (12, 16)
  const dim3 gF(FFN_DIM / 64, S_LEN / 128);   // (48, 16)
  const int attn_blocks = (NHEAD * (S_LEN / 16)) / 4;  // 384

  for (int b = 0; b < 4; ++b) {
    const int base = 3 + b * 16;
    ln_kernel<<<S_LEN, 256, 0, stream>>>(h, (const float*)d_in[base + 0],
                                         (const float*)d_in[base + 1], xb, nullptr);
    gemm_kernel<0><<<gE, 256, 0, stream>>>(xb, wq[b],  (const float*)d_in[base + 3],
                                           nullptr, qbuf, nullptr, S_LEN, EMB, EMB);
    gemm_kernel<0><<<gE, 256, 0, stream>>>(xb, wk[b],  (const float*)d_in[base + 5],
                                           nullptr, kbuf, nullptr, S_LEN, EMB, EMB);
    gemm_kernel<0><<<gE, 256, 0, stream>>>(xb, wvv[b], (const float*)d_in[base + 7],
                                           nullptr, vbuf, nullptr, S_LEN, EMB, EMB);
    attn_kernel<<<attn_blocks, 128, 0, stream>>>(qbuf, kbuf, vbuf, abuf);
    gemm_kernel<2><<<gE, 256, 0, stream>>>(abuf, wo[b], (const float*)d_in[base + 9],
                                           h, nullptr, h, S_LEN, EMB, EMB);
    ln_kernel<<<S_LEN, 256, 0, stream>>>(h, (const float*)d_in[base + 10],
                                         (const float*)d_in[base + 11], xb, nullptr);
    gemm_kernel<1><<<gF, 256, 0, stream>>>(xb, wd1[b], (const float*)d_in[base + 13],
                                           nullptr, midb, nullptr, S_LEN, FFN_DIM, EMB);
    gemm_kernel<2><<<gE, 256, 0, stream>>>(midb, wd2[b], (const float*)d_in[base + 15],
                                           h, nullptr, h, S_LEN, EMB, FFN_DIM);
  }

  ln_kernel<<<S_LEN, 256, 0, stream>>>(h, (const float*)d_in[67], (const float*)d_in[68], xb, hf);
  head_kernel<<<(VOCAB + 255) / 256, 256, 0, stream>>>(
      hf + (size_t)(S_LEN - 1) * EMB, (const float*)d_in[69], (const float*)d_in[70],
      (float*)d_out);
}
